// LIF_neuron_70592082477810
// MI455X (gfx1250) — matrix operand order repaired and checked by reading, not compiled, not run on hardware
//
#include <hip/hip_runtime.h>
#include <hip/hip_bf16.h>
#include <stdint.h>

// ---------------------------------------------------------------------------
// LIF neuron forward for MI455X (gfx1250, wave32, WMMA + async-to-LDS).
//
//   fcx = input @ W^T + b  (M=N=K=2048 fp32), then 15-step LIF recurrence.
//
// Pipeline:
//   1) split_f16_kernel: one pass over input/W, writes hi/lo f16 arrays to
//      workspace (fp32-accurate split: x = hi + lo, |err| ~ 2^-22 rel).
//   2) lif_wmma_async_kernel: double-buffered GLOBAL_LOAD_ASYNC_TO_LDS_B128
//      staging (ASYNCcnt), 3x v_wmma_f32_16x16x32_f16 per K-step
//      (Ah*Bh + Ah*Bl + Al*Bh, fp32 accum), LIF recurrence fused in epilogue.
//   Fallback (ws too small): round-1 fused-conversion kernel.
// ---------------------------------------------------------------------------

typedef __attribute__((ext_vector_type(16))) _Float16 v16h;
typedef __attribute__((ext_vector_type(8)))  _Float16 h8;
typedef __attribute__((ext_vector_type(4)))  _Float16 h4;
typedef __attribute__((ext_vector_type(8)))  float    v8f;

#define DECAY  0.2f
#define THRESH 0.2f

#define TILE_M 128
#define TILE_N 128
#define TILE_K 32
#define STRIDE_A 40   // async-path LDS row stride (halves): 80B, 16B-aligned
#define STRIDE_F 48   // fallback-path LDS row stride (halves)

// ---- CDNA5 async global->LDS copy (16B per lane) + asynccnt wait ----------
__device__ __forceinline__ void async_copy_b128(const _Float16* g, _Float16* l) {
#if __has_builtin(__builtin_amdgcn_global_load_async_to_lds_b128)
    // clang prototype: (vector_size(16) int AS1*, vector_size(16) int AS3*, Imm, Imm)
    typedef int v4i_vs __attribute__((vector_size(16)));
    typedef __attribute__((address_space(1))) v4i_vs* gasp_t;
    typedef __attribute__((address_space(3))) v4i_vs* lasp_t;
    __builtin_amdgcn_global_load_async_to_lds_b128(
        (gasp_t)(uintptr_t)g, (lasp_t)(uint32_t)(uintptr_t)l, 0, 0);
#else
    uint32_t lds32 = (uint32_t)(uintptr_t)l;
    asm volatile("global_load_async_to_lds_b128 %0, %1, off"
                 :: "v"(lds32), "v"((uint64_t)(uintptr_t)g) : "memory");
#endif
}

#if __has_builtin(__builtin_amdgcn_s_wait_asynccnt)
#define WAIT_ASYNC(n) __builtin_amdgcn_s_wait_asynccnt(n)
#else
#define WAIT_ASYNC(n) asm volatile("s_wait_asynccnt %0" :: "n"(n) : "memory")
#endif

// ---- WMMA fragment loaders (ISA 7.12.2 16-bit A/B layouts) ----------------
// A (16x32 MxK): lane<16: M=lane, e0..7->K0..7, e8..15->K16..23
//                lane>=16: M=lane-16, e0..7->K8..15, e8..15->K24..31
__device__ __forceinline__ v16h load_frag_a(const _Float16* rowp, int kb) {
    h8 x0 = *(const h8*)(rowp + kb);
    h8 x1 = *(const h8*)(rowp + kb + 16);
    v16h r;
#pragma unroll
    for (int e = 0; e < 8; ++e) { r[e] = x0[e]; r[e + 8] = x1[e]; }
    return r;
}
// B (32x16 KxN), K-contiguous per column: lane<16: N=lane, e0..7->K0..7, e8..15->K16..23
//                                         lane>=16: N=lane-16, e0..7->K8..15, e8..15->K24..31
__device__ __forceinline__ v16h load_frag_b(const _Float16* colp, int kb) {
    h8 x0 = *(const h8*)(colp + kb);
    h8 x1 = *(const h8*)(colp + kb + 16);
    v16h r;
#pragma unroll
    for (int e = 0; e < 8; ++e) { r[e] = x0[e]; r[e + 8] = x1[e]; }
    return r;
}

// ---- fused bias + LIF recurrence epilogue ---------------------------------
// C/D layout: elem r of v8f at lane l -> M = r + (l>=16 ? 8 : 0), N = l&15
__device__ __forceinline__ void lif_epilogue(const v8f (&acc)[2][4],
                                             const float* __restrict__ bias,
                                             float* __restrict__ out,
                                             int m0, int n0, int wm, int wn,
                                             int lane, int N, int win) {
    const int lh = lane & 15;
    const int mo = (lane & 16) ? 8 : 0;
#pragma unroll
    for (int j = 0; j < 4; ++j) {
        const int col = n0 + wn * 64 + j * 16 + lh;
        const float bb = bias[col];
#pragma unroll
        for (int i = 0; i < 2; ++i) {
            const v8f c = acc[i][j];
#pragma unroll
            for (int r = 0; r < 8; ++r) {
                const int mrow = m0 + wm * 32 + i * 16 + mo + r;
                const float fcx = c[r] + bb;
                float mem = 0.0f, spk = 0.0f;
                float* po = out + (size_t)mrow * (size_t)win * (size_t)N + col;
                for (int t = 0; t < win; ++t) {
                    mem = mem * DECAY * (1.0f - spk) + fcx;
                    spk = (mem > THRESH) ? 1.0f : 0.0f;
                    po[(size_t)t * N] = spk;
                }
            }
        }
    }
}

// ---------------------------------------------------------------------------
// Pre-pass: split fp32 -> (f16 hi, f16 lo), memory-bound one-shot transform.
// ---------------------------------------------------------------------------
__global__ __launch_bounds__(256)
void split_f16_kernel(const float* __restrict__ src,
                      _Float16* __restrict__ hi, _Float16* __restrict__ lo,
                      int n4) {  // n4 = n/4 float4 chunks
    int i = blockIdx.x * blockDim.x + threadIdx.x;
    if (i >= n4) return;
    const float4 v = *(const float4*)(src + (size_t)i * 4);
    const float f[4] = { v.x, v.y, v.z, v.w };
    h4 h, l;
#pragma unroll
    for (int e = 0; e < 4; ++e) {
        _Float16 x = (_Float16)f[e];
        h[e] = x;
        l[e] = (_Float16)(f[e] - (float)x);
    }
    *(h4*)(hi + (size_t)i * 4) = h;
    *(h4*)(lo + (size_t)i * 4) = l;
}

// ---------------------------------------------------------------------------
// Main GEMM+LIF: async-to-LDS double-buffered staging of pre-split f16 tiles.
// ---------------------------------------------------------------------------
#define TILE_HALVES (TILE_M * STRIDE_A)  // 5120 halves = 10240 B per tile

__global__ __launch_bounds__(256)
void lif_wmma_async_kernel(const _Float16* __restrict__ Ahi,  // [M,K]
                           const _Float16* __restrict__ Alo,
                           const _Float16* __restrict__ Bhi,  // [N,K]
                           const _Float16* __restrict__ Blo,
                           const float* __restrict__ bias,
                           float* __restrict__ out,
                           int M, int N, int K, int win) {
    // 2 buffers x 4 tiles x 128x40 halves = 80 KB
    __shared__ __align__(16) _Float16 smem[2 * 4 * TILE_HALVES];

    const int tid  = threadIdx.x;
    const int lane = tid & 31;
    const int w    = tid >> 5;
    const int wm   = w >> 1;
    const int wn   = w & 1;
    const int lh   = lane & 15;
    const int kbA  = (lane & 16) ? 8 : 0;
    const int kbB  = (lane & 16) ? 8 : 0;

    const int m0 = blockIdx.y * TILE_M;
    const int n0 = blockIdx.x * TILE_N;

    const _Float16* gsrc[4] = { Ahi + (size_t)m0 * K, Alo + (size_t)m0 * K,
                                Bhi + (size_t)n0 * K, Blo + (size_t)n0 * K };

    v8f acc[2][4];
#pragma unroll
    for (int i = 0; i < 2; ++i)
#pragma unroll
        for (int j = 0; j < 4; ++j) acc[i][j] = {};

    // issue one stage's async copies: 4 tiles x 512 16B chunks / 256 thr = 8/thr
    auto issue_stage = [&](int ks, int buf) {
#pragma unroll
        for (int a = 0; a < 4; ++a) {
            _Float16* lb = smem + (buf * 4 + a) * TILE_HALVES;
#pragma unroll
            for (int q = 0; q < 2; ++q) {
                const int c   = tid + q * 256;       // 512 chunks per tile
                const int row = c >> 2;
                const int h0  = (c & 3) * 8;
                async_copy_b128(gsrc[a] + (size_t)row * K + ks + h0,
                                lb + row * STRIDE_A + h0);
            }
        }
    };

    const int NS = K / TILE_K;
    issue_stage(0, 0);

    for (int s = 0; s < NS; ++s) {
        if (s + 1 < NS) {
            issue_stage((s + 1) * TILE_K, (s + 1) & 1);
            WAIT_ASYNC(8);          // stage s landed; stage s+1 in flight
        } else {
            WAIT_ASYNC(0);
        }
        __syncthreads();

        const _Float16* tAh = smem + (((s & 1) * 4) + 0) * TILE_HALVES;
        const _Float16* tAl = smem + (((s & 1) * 4) + 1) * TILE_HALVES;
        const _Float16* tBh = smem + (((s & 1) * 4) + 2) * TILE_HALVES;
        const _Float16* tBl = smem + (((s & 1) * 4) + 3) * TILE_HALVES;

        v16h aH[2], aL[2];
#pragma unroll
        for (int i = 0; i < 2; ++i) {
            const int row = wm * 32 + i * 16 + lh;
            aH[i] = load_frag_a(tAh + row * STRIDE_A, kbA);
            aL[i] = load_frag_a(tAl + row * STRIDE_A, kbA);
        }
#pragma unroll
        for (int j = 0; j < 4; ++j) {
            const int col = wn * 64 + j * 16 + lh;
            const v16h bH = load_frag_b(tBh + col * STRIDE_A, kbB);
            const v16h bL = load_frag_b(tBl + col * STRIDE_A, kbB);
#pragma unroll
            for (int i = 0; i < 2; ++i) {
                acc[i][j] = __builtin_amdgcn_wmma_f32_16x16x32_f16(
                    false, aH[i], false, bH, (short)0, acc[i][j], false, false);
                acc[i][j] = __builtin_amdgcn_wmma_f32_16x16x32_f16(
                    false, aH[i], false, bL, (short)0, acc[i][j], false, false);
                acc[i][j] = __builtin_amdgcn_wmma_f32_16x16x32_f16(
                    false, aL[i], false, bH, (short)0, acc[i][j], false, false);
            }
        }
        __syncthreads();   // all waves done reading buf before it is re-issued
    }

    lif_epilogue(acc, bias, out, m0, n0, wm, wn, lane, N, win);
}

// ---------------------------------------------------------------------------
// Fallback (ws too small): round-1 kernel with fused in-loop f32->f16 split.
// ---------------------------------------------------------------------------
__global__ __launch_bounds__(256)
void lif_wmma_fused_kernel(const float* __restrict__ inp,
                           const float* __restrict__ Wm,
                           const float* __restrict__ bias,
                           float* __restrict__ out,
                           int M, int N, int K, int win) {
    __shared__ __align__(16) _Float16 Ah[TILE_M][STRIDE_F];
    __shared__ __align__(16) _Float16 Al[TILE_M][STRIDE_F];
    __shared__ __align__(16) _Float16 Bh[TILE_N][STRIDE_F];
    __shared__ __align__(16) _Float16 Bl[TILE_N][STRIDE_F];

    const int tid  = threadIdx.x;
    const int lane = tid & 31;
    const int w    = tid >> 5;
    const int wm   = w >> 1;
    const int wn   = w & 1;
    const int lh   = lane & 15;
    const int kbA  = (lane & 16) ? 8 : 0;
    const int kbB  = (lane & 16) ? 8 : 0;

    const int m0 = blockIdx.y * TILE_M;
    const int n0 = blockIdx.x * TILE_N;

    v8f acc[2][4];
#pragma unroll
    for (int i = 0; i < 2; ++i)
#pragma unroll
        for (int j = 0; j < 4; ++j) acc[i][j] = {};

    for (int ks = 0; ks < K; ks += TILE_K) {
        float4 va[4], vb[4];
#pragma unroll
        for (int i = 0; i < 4; ++i) {
            const int c   = tid + 256 * i;
            const int row = c >> 3;
            const int c4  = (c & 7) * 4;
            va[i] = *(const float4*)(inp + (size_t)(m0 + row) * K + ks + c4);
            vb[i] = *(const float4*)(Wm  + (size_t)(n0 + row) * K + ks + c4);
        }
        __syncthreads();
#pragma unroll
        for (int i = 0; i < 4; ++i) {
            const int c   = tid + 256 * i;
            const int row = c >> 3;
            const int c4  = (c & 7) * 4;
            const float fa[4] = { va[i].x, va[i].y, va[i].z, va[i].w };
            const float fb[4] = { vb[i].x, vb[i].y, vb[i].z, vb[i].w };
            h4 ah, al, bh, bl;
#pragma unroll
            for (int e = 0; e < 4; ++e) {
                _Float16 h = (_Float16)fa[e];
                ah[e] = h; al[e] = (_Float16)(fa[e] - (float)h);
                h = (_Float16)fb[e];
                bh[e] = h; bl[e] = (_Float16)(fb[e] - (float)h);
            }
            *(h4*)&Ah[row][c4] = ah;  *(h4*)&Al[row][c4] = al;
            *(h4*)&Bh[row][c4] = bh;  *(h4*)&Bl[row][c4] = bl;
        }
        __syncthreads();

        v16h aH[2], aL[2];
#pragma unroll
        for (int i = 0; i < 2; ++i) {
            const int row = wm * 32 + i * 16 + lh;
            aH[i] = load_frag_a(&Ah[row][0], kbA);
            aL[i] = load_frag_a(&Al[row][0], kbA);
        }
#pragma unroll
        for (int j = 0; j < 4; ++j) {
            const int col = wn * 64 + j * 16 + lh;
            const v16h bH = load_frag_b(&Bh[col][0], kbB);
            const v16h bL = load_frag_b(&Bl[col][0], kbB);
#pragma unroll
            for (int i = 0; i < 2; ++i) {
                acc[i][j] = __builtin_amdgcn_wmma_f32_16x16x32_f16(
                    false, aH[i], false, bH, (short)0, acc[i][j], false, false);
                acc[i][j] = __builtin_amdgcn_wmma_f32_16x16x32_f16(
                    false, aH[i], false, bL, (short)0, acc[i][j], false, false);
                acc[i][j] = __builtin_amdgcn_wmma_f32_16x16x32_f16(
                    false, aL[i], false, bH, (short)0, acc[i][j], false, false);
            }
        }
    }
    lif_epilogue(acc, bias, out, m0, n0, wm, wn, lane, N, win);
}

// ---------------------------------------------------------------------------
extern "C" void kernel_launch(void* const* d_in, const int* in_sizes, int n_in,
                              void* d_out, int out_size, void* d_ws, size_t ws_size,
                              hipStream_t stream) {
    const float* inp  = (const float*)d_in[0];
    const float* Wm   = (const float*)d_in[1];
    const float* bias = (const float*)d_in[2];
    float*       out  = (float*)d_out;

    const int OUT = in_sizes[2];            // N
    const int IN  = in_sizes[1] / OUT;      // K
    const int Bm  = in_sizes[0] / IN;       // M
    const int win = out_size / (Bm * OUT);  // time steps

    const size_t MK = (size_t)Bm * IN;
    const size_t NK = (size_t)OUT * IN;
    const size_t need = (2 * MK + 2 * NK) * sizeof(_Float16);

    dim3 grid(OUT / TILE_N, Bm / TILE_M);
    dim3 block(256);

    if (d_ws != nullptr && ws_size >= need) {
        _Float16* wsh = (_Float16*)d_ws;
        _Float16* Ahi = wsh;
        _Float16* Alo = wsh + MK;
        _Float16* Bhi = wsh + 2 * MK;
        _Float16* Blo = wsh + 2 * MK + NK;

        const int a4 = (int)(MK / 4), b4 = (int)(NK / 4);
        split_f16_kernel<<<(a4 + 255) / 256, 256, 0, stream>>>(inp, Ahi, Alo, a4);
        split_f16_kernel<<<(b4 + 255) / 256, 256, 0, stream>>>(Wm,  Bhi, Blo, b4);
        lif_wmma_async_kernel<<<grid, block, 0, stream>>>(
            Ahi, Alo, Bhi, Blo, bias, out, Bm, OUT, IN, win);
    } else {
        lif_wmma_fused_kernel<<<grid, block, 0, stream>>>(
            inp, Wm, bias, out, Bm, OUT, IN, win);
    }
}
